// DKF_4398046511160
// MI455X (gfx1250) — compile-verified, matrix operand order checked by reading, not run on hardware
//
#include <hip/hip_runtime.h>
#include <hip/hip_bf16.h>

// ---------------- WMMA fragment helpers (gfx1250, wave32) ----------------
typedef __attribute__((ext_vector_type(16))) _Float16 v16h;
typedef __attribute__((ext_vector_type(8)))  float    v8f;

#define B_SZ 256
#define T_SZ 512
#define D_SZ 128
#define Z_SZ 128
#define TR_SZ 256
#define EM_SZ 256
#define RH_SZ 512

__device__ inline v8f wmma_f16(v16h a, v16h b, v8f c) {
  // D = A(16x32 f16) * B(32x16 f16) + C(16x16 f32)
  return __builtin_amdgcn_wmma_f32_16x16x32_f16(
      /*neg_a=*/false, a, /*neg_b=*/false, b,
      /*c_mod=*/(short)0, c, /*reuse_a=*/false, /*reuse_b=*/false);
}

// A fragment: 16x32 f16 tile from row-major [16][ld] at p0.
// lane: rows 0..15 twice; kb = (lane>=16)?8:0 ; halves K=kb..kb+7, kb+16..kb+23
__device__ inline v16h ld_frag_h(const _Float16* p0, int ld, int lane) {
  int r  = lane & 15;
  int kb = (lane >> 4) << 3;
  const _Float16* p = p0 + (size_t)r * ld + kb;
  v16h a;
#pragma unroll
  for (int i = 0; i < 8; ++i) { a[i] = p[i]; a[i + 8] = p[i + 16]; }
  return a;
}

// Same, but apply ReLU on load (for relu(prop) @ W_lv)
__device__ inline v16h ld_frag_h_relu(const _Float16* p0, int ld, int lane) {
  int r  = lane & 15;
  int kb = (lane >> 4) << 3;
  const _Float16* p = p0 + (size_t)r * ld + kb;
  v16h a;
#pragma unroll
  for (int i = 0; i < 8; ++i) {
    float v0 = (float)p[i];      a[i]     = (_Float16)fmaxf(v0, 0.f);
    float v1 = (float)p[i + 16]; a[i + 8] = (_Float16)fmaxf(v1, 0.f);
  }
  return a;
}

// A fragment from f32 row-major source (converted on the fly)
__device__ inline v16h ld_frag_f32(const float* p0, size_t ld, int lane) {
  int r  = lane & 15;
  int kb = (lane >> 4) << 3;
  const float* p = p0 + (size_t)r * ld + kb;
  v16h a;
#pragma unroll
  for (int i = 0; i < 8; ++i) {
    a[i]     = (_Float16)p[i];
    a[i + 8] = (_Float16)p[i + 16];
  }
  return a;
}

// accumulate [16 x 16] tile: A is LDS f16 [16][lda], WT is global f16 [N][ldw]
__device__ inline v8f gemm_hh(const _Float16* A, int lda, const _Float16* WT,
                              int ldw, int K, int lane) {
  v8f acc = {0.f, 0.f, 0.f, 0.f, 0.f, 0.f, 0.f, 0.f};
  for (int kt = 0; kt < K; kt += 32)
    acc = wmma_f16(ld_frag_h(A + kt, lda, lane), ld_frag_h(WT + kt, ldw, lane), acc);
  return acc;
}

__device__ inline v8f gemm_hh_reluA(const _Float16* A, int lda, const _Float16* WT,
                                    int ldw, int K, int lane) {
  v8f acc = {0.f, 0.f, 0.f, 0.f, 0.f, 0.f, 0.f, 0.f};
  for (int kt = 0; kt < K; kt += 32)
    acc = wmma_f16(ld_frag_h_relu(A + kt, lda, lane), ld_frag_h(WT + kt, ldw, lane), acc);
  return acc;
}

// C/D layout: VGPR i -> row (i + 8*(lane>>4)), col (lane&15)
__device__ inline void st_h16(v8f c, _Float16* out, int ld, int n0,
                              const float* bias, int lane, int act) {
  int n  = n0 + (lane & 15);
  int mo = (lane >> 4) * 8;
  float bv = bias[n];
#pragma unroll
  for (int i = 0; i < 8; ++i) {
    float v = c[i] + bv;
    if (act == 1)      v = fmaxf(v, 0.f);
    else if (act == 2) v = 1.f / (1.f + __expf(-v));
    out[(i + mo) * ld + n] = (_Float16)v;
  }
}

__device__ inline void st_f32(v8f c, float* out, int n0, const float* bias, int lane) {
  int n  = n0 + (lane & 15);
  int mo = (lane >> 4) * 8;
  float bv = bias[n];
#pragma unroll
  for (int i = 0; i < 8; ++i) out[(i + mo) * Z_SZ + n] = c[i] + bv;
}

// ---------------- prep: weight transpose + fp32 -> f16 ----------------
__global__ void dkf_transpose_f16(const float* __restrict__ W,
                                  _Float16* __restrict__ WT, int K, int N) {
  int idx = blockIdx.x * blockDim.x + threadIdx.x;
  if (idx < K * N) {
    int k = idx / N, n = idx - k * N;
    WT[(size_t)n * K + k] = (_Float16)W[idx];
  }
}

__global__ void dkf_zero(float* accum) { accum[0] = 0.f; accum[1] = 0.f; }

// ---------------- RNN recurrence (input proj fused) ----------------
// 16 blocks x 256 threads. Block owns 16 batch rows; serial loop over T.
// h_t = relu(x_t @ Wih + h_{t-1} @ Whh + bih + bhh)
__global__ __launch_bounds__(256) void dkf_rnn_kernel(
    const float* __restrict__ x, const _Float16* __restrict__ WihT,
    const _Float16* __restrict__ WhhT, const float* __restrict__ bih,
    const float* __restrict__ bhh, const float* __restrict__ h0,
    _Float16* __restrict__ rnn_out) {
  __shared__ _Float16 hbuf[2][16 * RH_SZ];
  const int b0 = blockIdx.x * 16;
  const int tid = threadIdx.x, lane = tid & 31, wave = tid >> 5;

  for (int i = tid; i < 16 * RH_SZ; i += 256)
    hbuf[0][i] = (_Float16)h0[i & (RH_SZ - 1)];
  __syncthreads();

  for (int t = 0; t < T_SZ; ++t) {
    const int cur = t & 1, nxt = cur ^ 1;
    const float* xbase = x + ((size_t)b0 * T_SZ + t) * D_SZ; // row stride T*D
    for (int nt = wave; nt < RH_SZ / 16; nt += 8) {
      v8f acc = {0.f, 0.f, 0.f, 0.f, 0.f, 0.f, 0.f, 0.f};
      const _Float16* wrow = WhhT + (size_t)nt * 16 * RH_SZ;
      for (int kt = 0; kt < RH_SZ; kt += 32) {
        __builtin_prefetch(wrow + kt + 512, 0, 1); // stream Whh via L2
        acc = wmma_f16(ld_frag_h(&hbuf[cur][kt], RH_SZ, lane),
                       ld_frag_h(wrow + kt, RH_SZ, lane), acc);
      }
      const _Float16* wih = WihT + (size_t)nt * 16 * D_SZ;
      for (int kt = 0; kt < D_SZ; kt += 32)
        acc = wmma_f16(ld_frag_f32(xbase + kt, (size_t)T_SZ * D_SZ, lane),
                       ld_frag_h(wih + kt, D_SZ, lane), acc);
      int n  = nt * 16 + (lane & 15);
      int mo = (lane >> 4) * 8;
      float bias = bih[n] + bhh[n];
#pragma unroll
      for (int i = 0; i < 8; ++i) {
        int m = i + mo;
        float v = fmaxf(acc[i] + bias, 0.f);
        _Float16 hv = (_Float16)v;
        hbuf[nxt][m * RH_SZ + n] = hv;
        rnn_out[(size_t)t * (B_SZ * RH_SZ) + (size_t)(b0 + m) * RH_SZ + n] = hv;
      }
    }
    __syncthreads();
  }
}

// ---------------- latent scan (transition + combiner + emitter + losses) ---
__global__ __launch_bounds__(256) void dkf_latent_kernel(
    const _Float16* __restrict__ rnn_out, const float* __restrict__ y,
    const float* __restrict__ eps_comb, const float* __restrict__ eps_emit,
    const _Float16* g1T, const float* g1b, const _Float16* g2T, const float* g2b,
    const _Float16* p1T, const float* p1b, const _Float16* p2T, const float* p2b,
    const _Float16* trmuT, const float* trmub, const _Float16* trlvT, const float* trlvb,
    const _Float16* cbhT, const float* cbhb, const _Float16* cbmuT, const float* cbmub,
    const _Float16* cblvT, const float* cblvb,
    const _Float16* e1T, const float* e1b, const _Float16* e2T, const float* e2b,
    const _Float16* emuT, const float* emub, const float* __restrict__ em_logvar,
    const float* __restrict__ zq0, float* accum) {
  __shared__ _Float16 zH[16 * Z_SZ];      // current z (A operand)
  __shared__ _Float16 bufH[16 * RH_SZ];   // g1/p1/hc; h1 in [0,4096), h2 in [4096,8192)
  __shared__ _Float16 gateH[16 * Z_SZ];
  __shared__ _Float16 propH[16 * Z_SZ];   // raw prop (relu applied in A-load)
  __shared__ float prmuF[16 * Z_SZ];      // zmu then pr_mu
  __shared__ float prlvF[16 * Z_SZ];
  __shared__ float muF[16 * Z_SZ];
  __shared__ float lvF[16 * Z_SZ];

  const int b0 = blockIdx.x * 16;
  const int tid = threadIdx.x, lane = tid & 31, wave = tid >> 5;
  float klAcc = 0.f, recAcc = 0.f;

  for (int i = tid; i < 16 * Z_SZ; i += 256)
    zH[i] = (_Float16)zq0[i & (Z_SZ - 1)];
  __syncthreads();

  for (int t = 0; t < T_SZ; ++t) {
    // 1: g1 = relu(z @ g1W + b)  [16 x 256]
    for (int nt = wave; nt < TR_SZ / 16; nt += 8)
      st_h16(gemm_hh(zH, Z_SZ, g1T + (size_t)nt * 16 * Z_SZ, Z_SZ, Z_SZ, lane),
             bufH, TR_SZ, nt * 16, g1b, lane, 1);
    __syncthreads();
    // 2: gate = sigmoid(g1 @ g2W + b)  [16 x 128]
    for (int nt = wave; nt < Z_SZ / 16; nt += 8)
      st_h16(gemm_hh(bufH, TR_SZ, g2T + (size_t)nt * 16 * TR_SZ, TR_SZ, TR_SZ, lane),
             gateH, Z_SZ, nt * 16, g2b, lane, 2);
    __syncthreads();
    // 3: p1 = relu(z @ p1W + b)
    for (int nt = wave; nt < TR_SZ / 16; nt += 8)
      st_h16(gemm_hh(zH, Z_SZ, p1T + (size_t)nt * 16 * Z_SZ, Z_SZ, Z_SZ, lane),
             bufH, TR_SZ, nt * 16, p1b, lane, 1);
    __syncthreads();
    // 4: prop = p1 @ p2W + b  (raw, f16)
    for (int nt = wave; nt < Z_SZ / 16; nt += 8)
      st_h16(gemm_hh(bufH, TR_SZ, p2T + (size_t)nt * 16 * TR_SZ, TR_SZ, TR_SZ, lane),
             propH, Z_SZ, nt * 16, p2b, lane, 0);
    __syncthreads();
    // 5: zmu = z @ trmuW + b ; pr_lv = relu(prop) @ trlvW + b
    for (int nt = wave; nt < 16; nt += 8) {
      if (nt < 8)
        st_f32(gemm_hh(zH, Z_SZ, trmuT + (size_t)nt * 16 * Z_SZ, Z_SZ, Z_SZ, lane),
               prmuF, nt * 16, trmub, lane);
      else {
        int q = nt - 8;
        st_f32(gemm_hh_reluA(propH, Z_SZ, trlvT + (size_t)q * 16 * Z_SZ, Z_SZ, Z_SZ, lane),
               prlvF, q * 16, trlvb, lane);
      }
    }
    __syncthreads();
    // 6: pr_mu = (1-gate)*zmu + gate*prop  (elementwise, in place)
    for (int i = tid; i < 16 * Z_SZ; i += 256) {
      float g = (float)gateH[i], pp = (float)propH[i];
      prmuF[i] = (1.f - g) * prmuF[i] + g * pp;
    }
    // 7: hc = 0.5*(tanh(z @ cbhW + b) + h_rnn)  [16 x 512]
    for (int nt = wave; nt < RH_SZ / 16; nt += 8) {
      v8f c = gemm_hh(zH, Z_SZ, cbhT + (size_t)nt * 16 * Z_SZ, Z_SZ, Z_SZ, lane);
      int n = nt * 16 + (lane & 15), mo = (lane >> 4) * 8;
      float bv = cbhb[n];
#pragma unroll
      for (int i = 0; i < 8; ++i) {
        int m = i + mo;
        float v  = tanhf(c[i] + bv);
        float hr = (float)rnn_out[(size_t)t * (B_SZ * RH_SZ) +
                                  (size_t)(b0 + m) * RH_SZ + n];
        bufH[m * RH_SZ + n] = (_Float16)(0.5f * (v + hr));
      }
    }
    __syncthreads();
    // 8: mu = hc @ cbmuW + b ; lv = hc @ cblvW + b
    for (int nt = wave; nt < 16; nt += 8) {
      if (nt < 8)
        st_f32(gemm_hh(bufH, RH_SZ, cbmuT + (size_t)nt * 16 * RH_SZ, RH_SZ, RH_SZ, lane),
               muF, nt * 16, cbmub, lane);
      else {
        int q = nt - 8;
        st_f32(gemm_hh(bufH, RH_SZ, cblvT + (size_t)q * 16 * RH_SZ, RH_SZ, RH_SZ, lane),
               lvF, q * 16, cblvb, lane);
      }
    }
    __syncthreads();
    // 9: reparam z_t + KL accumulation (elementwise)
    for (int i = tid; i < 16 * Z_SZ; i += 256) {
      int m = i >> 7, n = i & (Z_SZ - 1), bb = b0 + m;
      float mu = muF[i], lv = lvF[i];
      float ec = eps_comb[(size_t)t * (B_SZ * Z_SZ) + (size_t)bb * Z_SZ + n];
      float z  = mu + ec * __expf(0.5f * lv);
      zH[i] = (_Float16)z;
      float pm = prmuF[i], pl = prlvF[i], d = mu - pm;
      klAcc += 0.5f * (pl - lv + (__expf(lv) + d * d) * __expf(-pl) - 1.f);
    }
    __syncthreads();
    // 10: h1 = relu(z_t @ e1W + b)  -> bufH[0,4096)
    for (int nt = wave; nt < EM_SZ / 16; nt += 8)
      st_h16(gemm_hh(zH, Z_SZ, e1T + (size_t)nt * 16 * Z_SZ, Z_SZ, Z_SZ, lane),
             bufH, EM_SZ, nt * 16, e1b, lane, 1);
    __syncthreads();
    // 11: h2 = relu(h1 @ e2W + b)  -> bufH[4096,8192)
    for (int nt = wave; nt < EM_SZ / 16; nt += 8)
      st_h16(gemm_hh(bufH, EM_SZ, e2T + (size_t)nt * 16 * EM_SZ, EM_SZ, EM_SZ, lane),
             bufH + 16 * EM_SZ, EM_SZ, nt * 16, e2b, lane, 1);
    __syncthreads();
    // 12: x_mu = h2 @ emuW + b ; x_t = x_mu + eps*exp(0.5*logvar); rec += (x_t-y)^2
    for (int nt = wave; nt < D_SZ / 16; nt += 8) {
      v8f c = gemm_hh(bufH + 16 * EM_SZ, EM_SZ, emuT + (size_t)nt * 16 * EM_SZ,
                      EM_SZ, EM_SZ, lane);
      int n = nt * 16 + (lane & 15), mo = (lane >> 4) * 8;
      float bv = emub[n];
      float sc = __expf(0.5f * em_logvar[n]);
#pragma unroll
      for (int i = 0; i < 8; ++i) {
        int bb = b0 + i + mo;
        float ee = eps_emit[(size_t)t * (B_SZ * D_SZ) + (size_t)bb * D_SZ + n];
        float xt = c[i] + bv + ee * sc;
        float dy = xt - y[((size_t)bb * T_SZ + t) * D_SZ + n];
        recAcc += dy * dy;
      }
    }
    __syncthreads();
  }
  atomicAdd(&accum[0], recAcc);
  atomicAdd(&accum[1], klAcc);
}

__global__ void dkf_finalize(const float* accum, float* out) {
  out[0] = accum[0] / ((float)T_SZ * (float)B_SZ * (float)D_SZ);
  out[1] = accum[1] / ((float)T_SZ * (float)B_SZ);
}

// ---------------- host launch ----------------
extern "C" void kernel_launch(void* const* d_in, const int* in_sizes, int n_in,
                              void* d_out, int out_size, void* d_ws, size_t ws_size,
                              hipStream_t stream) {
  const float* x        = (const float*)d_in[0];
  const float* y        = (const float*)d_in[1];
  const float* eps_comb = (const float*)d_in[2];
  const float* eps_emit = (const float*)d_in[3];
  const float* rnn_Wih  = (const float*)d_in[4];
  const float* rnn_Whh  = (const float*)d_in[5];
  const float* rnn_bih  = (const float*)d_in[6];
  const float* rnn_bhh  = (const float*)d_in[7];
  const float* h0       = (const float*)d_in[8];
  const float* zq0      = (const float*)d_in[9];
  const float* tr_g1_W  = (const float*)d_in[10];
  const float* tr_g1_b  = (const float*)d_in[11];
  const float* tr_g2_W  = (const float*)d_in[12];
  const float* tr_g2_b  = (const float*)d_in[13];
  const float* tr_p1_W  = (const float*)d_in[14];
  const float* tr_p1_b  = (const float*)d_in[15];
  const float* tr_p2_W  = (const float*)d_in[16];
  const float* tr_p2_b  = (const float*)d_in[17];
  const float* tr_mu_W  = (const float*)d_in[18];
  const float* tr_mu_b  = (const float*)d_in[19];
  const float* tr_lv_W  = (const float*)d_in[20];
  const float* tr_lv_b  = (const float*)d_in[21];
  const float* cb_h_W   = (const float*)d_in[22];
  const float* cb_h_b   = (const float*)d_in[23];
  const float* cb_mu_W  = (const float*)d_in[24];
  const float* cb_mu_b  = (const float*)d_in[25];
  const float* cb_lv_W  = (const float*)d_in[26];
  const float* cb_lv_b  = (const float*)d_in[27];
  const float* em_h1_W  = (const float*)d_in[28];
  const float* em_h1_b  = (const float*)d_in[29];
  const float* em_h2_W  = (const float*)d_in[30];
  const float* em_h2_b  = (const float*)d_in[31];
  const float* em_mu_W  = (const float*)d_in[32];
  const float* em_mu_b  = (const float*)d_in[33];
  const float* em_lvp   = (const float*)d_in[34];

  // workspace layout (element offsets in f16 halves for the weight pool)
  char* ws = (char*)d_ws;
  _Float16* hpool = (_Float16*)ws;
  _Float16* WihT  = hpool + 0;        // [512][128]
  _Float16* WhhT  = hpool + 65536;    // [512][512]
  _Float16* g1T   = hpool + 327680;   // [256][128]
  _Float16* g2T   = hpool + 360448;   // [128][256]
  _Float16* p1T   = hpool + 393216;   // [256][128]
  _Float16* p2T   = hpool + 425984;   // [128][256]
  _Float16* trmuT = hpool + 458752;   // [128][128]
  _Float16* trlvT = hpool + 475136;   // [128][128]
  _Float16* cbhT  = hpool + 491520;   // [512][128]
  _Float16* cbmuT = hpool + 557056;   // [128][512]
  _Float16* cblvT = hpool + 622592;   // [128][512]
  _Float16* e1T   = hpool + 688128;   // [256][128]
  _Float16* e2T   = hpool + 720896;   // [256][256]
  _Float16* emuT  = hpool + 786432;   // [128][256]
  const size_t OFF_RNN = 1638400;                       // bytes
  _Float16* rnn_out = (_Float16*)(ws + OFF_RNN);        // [T][B][RH] f16 = 128MB
  float* accum = (float*)(ws + OFF_RNN + (size_t)T_SZ * B_SZ * RH_SZ * 2);

  dkf_zero<<<1, 1, 0, stream>>>(accum);

  auto tr = [&](const float* W, _Float16* WT, int K, int N) {
    int total = K * N;
    dkf_transpose_f16<<<(total + 255) / 256, 256, 0, stream>>>(W, WT, K, N);
  };
  tr(rnn_Wih, WihT, D_SZ, RH_SZ);
  tr(rnn_Whh, WhhT, RH_SZ, RH_SZ);
  tr(tr_g1_W, g1T, Z_SZ, TR_SZ);
  tr(tr_g2_W, g2T, TR_SZ, Z_SZ);
  tr(tr_p1_W, p1T, Z_SZ, TR_SZ);
  tr(tr_p2_W, p2T, TR_SZ, Z_SZ);
  tr(tr_mu_W, trmuT, Z_SZ, Z_SZ);
  tr(tr_lv_W, trlvT, Z_SZ, Z_SZ);
  tr(cb_h_W, cbhT, Z_SZ, RH_SZ);
  tr(cb_mu_W, cbmuT, RH_SZ, Z_SZ);
  tr(cb_lv_W, cblvT, RH_SZ, Z_SZ);
  tr(em_h1_W, e1T, Z_SZ, EM_SZ);
  tr(em_h2_W, e2T, EM_SZ, EM_SZ);
  tr(em_mu_W, emuT, EM_SZ, D_SZ);

  dkf_rnn_kernel<<<B_SZ / 16, 256, 0, stream>>>(x, WihT, WhhT, rnn_bih, rnn_bhh,
                                                h0, rnn_out);

  dkf_latent_kernel<<<B_SZ / 16, 256, 0, stream>>>(
      rnn_out, y, eps_comb, eps_emit,
      g1T, tr_g1_b, g2T, tr_g2_b, p1T, tr_p1_b, p2T, tr_p2_b,
      trmuT, tr_mu_b, trlvT, tr_lv_b,
      cbhT, cb_h_b, cbmuT, cb_mu_b, cblvT, cb_lv_b,
      e1T, em_h1_b, e2T, em_h2_b, emuT, em_mu_b, em_lvp, zq0, accum);

  dkf_finalize<<<1, 1, 0, stream>>>(accum, (float*)d_out);
}
